// Attention_55894704390742
// MI455X (gfx1250) — matrix operand order repaired and checked by reading, not compiled, not run on hardware
//
#include <hip/hip_runtime.h>

// Problem constants
#define BB   2
#define SS   2048
#define NXX  1024
#define NST  1024
#define NH   16
#define HD   64
#define ROWS (BB * SS)      // 4096
#define NC   (3 * NST)      // 3072

typedef __attribute__((ext_vector_type(4)))  __bf16 bf16x4;
typedef __attribute__((ext_vector_type(8)))  __bf16 bf16x8;
typedef __attribute__((ext_vector_type(16))) __bf16 bf16x16;
typedef __attribute__((ext_vector_type(8)))  float  f32x8;

__device__ __forceinline__ bf16x8 ld8(const __bf16* p) {
    return *reinterpret_cast<const bf16x8*>(p);
}
__device__ __forceinline__ bf16x16 cat16(bf16x8 a, bf16x8 b) {
    return __builtin_shufflevector(a, b, 0,1,2,3,4,5,6,7,8,9,10,11,12,13,14,15);
}
__device__ __forceinline__ f32x8 wmma_bf16(bf16x16 a, bf16x16 b, f32x8 c) {
    return __builtin_amdgcn_wmma_f32_16x16x32_bf16(false, a, false, b, (short)0, c,
                                                   false, false);
}

// ---------------------------------------------------------------- converters
__global__ void cvt_f32_to_bf16_x4(const float* __restrict__ in,
                                   __bf16* __restrict__ out, int n4) {
    int i = blockIdx.x * blockDim.x + threadIdx.x;
    if (i >= n4) return;
    float4 v = reinterpret_cast<const float4*>(in)[i];
    bf16x4 o;
    o.x = (__bf16)v.x; o.y = (__bf16)v.y; o.z = (__bf16)v.z; o.w = (__bf16)v.w;
    *reinterpret_cast<bf16x4*>(out + 4 * i) = o;
}

// W[K,N] f32 row-major -> Wt[N,K] bf16 row-major (LDS 32x32 tile transpose)
__global__ void transpose_to_bf16(const float* __restrict__ W,
                                  __bf16* __restrict__ Wt, int K, int N) {
    __shared__ float tile[32][33];
    int n0 = blockIdx.x * 32, k0 = blockIdx.y * 32;
    int tx = threadIdx.x, ty = threadIdx.y;       // block (32, 8)
    for (int r = ty; r < 32; r += 8)
        tile[r][tx] = W[(size_t)(k0 + r) * N + (n0 + tx)];
    __syncthreads();
    for (int r = ty; r < 32; r += 8)
        Wt[(size_t)(n0 + r) * K + (k0 + tx)] = (__bf16)tile[tx][r];
}

// ------------------------------------------------------- 32x64 WMMA GEMM tile
// C[32,64] += A[32,K] * Bt[64,K]^T  — each B tile feeds 2 WMMAs (register reuse)
__device__ __forceinline__ void gemm_tile_32x64(
    const __bf16* __restrict__ A, const __bf16* __restrict__ Bt,
    int K, int rowBase, int colBase, int lane, f32x8 acc[8]) {
    const int sel  = (lane & 16) ? 8 : 0;    // unused; A and B both use nsel
    const int nsel = (lane & 16) ? 16 : 0;   // B-operand K sub-select
    const int nl   = lane & 15;
    const __bf16* aRow0 = A + (size_t)(rowBase + nl) * K;
    const __bf16* aRow1 = A + (size_t)(rowBase + 16 + nl) * K;
    const __bf16* b0 = Bt + (size_t)(colBase +  0 + nl) * K + nsel;
    const __bf16* b1 = Bt + (size_t)(colBase + 16 + nl) * K + nsel;
    const __bf16* b2 = Bt + (size_t)(colBase + 32 + nl) * K + nsel;
    const __bf16* b3 = Bt + (size_t)(colBase + 48 + nl) * K + nsel;
    for (int k = 0; k < K; k += 32) {
        bf16x16 av0 = cat16(ld8(aRow0 + k + nsel), ld8(aRow0 + k + nsel + 8));
        bf16x16 av1 = cat16(ld8(aRow1 + k + nsel), ld8(aRow1 + k + nsel + 8));
        bf16x16 bv;
        bv = cat16(ld8(b0 + k), ld8(b0 + k + 8));
        acc[0] = wmma_bf16(av0, bv, acc[0]);
        acc[4] = wmma_bf16(av1, bv, acc[4]);
        bv = cat16(ld8(b1 + k), ld8(b1 + k + 8));
        acc[1] = wmma_bf16(av0, bv, acc[1]);
        acc[5] = wmma_bf16(av1, bv, acc[5]);
        bv = cat16(ld8(b2 + k), ld8(b2 + k + 8));
        acc[2] = wmma_bf16(av0, bv, acc[2]);
        acc[6] = wmma_bf16(av1, bv, acc[6]);
        bv = cat16(ld8(b3 + k), ld8(b3 + k + 8));
        acc[3] = wmma_bf16(av0, bv, acc[3]);
        acc[7] = wmma_bf16(av1, bv, acc[7]);
    }
}

// -------------------------------------------------------------- QKV GEMM
// Xb[4096,1024] @ Wc[1024,3072] + b_c  -> Qb[b,h,s,d]*0.125 | Kb[b,h,s,d] | Vt[b,h,d,s]
// Block = 8 waves x 32 rows (256 rows) sharing one 64-col B panel.
__global__ void __launch_bounds__(256) qkv_gemm(
    const __bf16* __restrict__ Xb, const __bf16* __restrict__ WcT,
    const float* __restrict__ bc,
    __bf16* __restrict__ Qb, __bf16* __restrict__ Kb, __bf16* __restrict__ Vt) {
    int lane = threadIdx.x & 31, wave = threadIdx.x >> 5;
    int rb = blockIdx.x / (NC / 64);          // 16 row blocks of 256
    int ct = blockIdx.x % (NC / 64);          // 48 col panels
    int rowBase = rb * 256 + wave * 32, colBase = ct * 64;
    f32x8 zero = {0.f,0.f,0.f,0.f,0.f,0.f,0.f,0.f};
    f32x8 acc[8] = {zero, zero, zero, zero, zero, zero, zero, zero};
    gemm_tile_32x64(Xb, WcT, NXX, rowBase, colBase, lane, acc);

    int msel = (lane & 16) ? 8 : 0;
    for (int t = 0; t < 4; ++t) {
        int col = colBase + t * 16 + (lane & 15);
        int which = col >> 10, n = col & 1023;
        int h = n >> 6, d = n & 63;
        float bias = bc[col];
        for (int g = 0; g < 2; ++g) {
            for (int v = 0; v < 8; ++v) {
                int row = rowBase + g * 16 + v + msel;
                int bi = row >> 11, s = row & (SS - 1);
                float val = acc[g * 4 + t][v] + bias;
                size_t bh = (size_t)(bi * NH + h);
                if (which == 0)
                    Qb[(bh * SS + s) * HD + d] = (__bf16)(val * 0.125f); // 1/sqrt(64)
                else if (which == 1)
                    Kb[(bh * SS + s) * HD + d] = (__bf16)val;
                else
                    Vt[(bh * HD + d) * SS + s] = (__bf16)val;
            }
        }
    }
}

// ------------------------------------------------------------- attention
// One workgroup (8 waves) per (b*h, 16-query tile). Scores in dynamic LDS.
#define ATTN_SMEM ((16 * SS + 2 * 16 * HD) * 4)

__global__ void __launch_bounds__(256) attn_kernel(
    const __bf16* __restrict__ Qb, const __bf16* __restrict__ Kb,
    const __bf16* __restrict__ Vt, __bf16* __restrict__ Ab) {
    extern __shared__ float smem[];
    float* scores = smem;            // [16][2048]
    float* red    = smem + 16 * SS;  // [2][16][64]

    int qt = blockIdx.x & 127;       // S/16 tiles
    int bh = blockIdx.x >> 7;        // b*16 + h
    int lane = threadIdx.x & 31, wave = threadIdx.x >> 5;
    const __bf16* Q  = Qb + (size_t)bh * SS * HD;
    const __bf16* Kp = Kb + (size_t)bh * SS * HD;
    const __bf16* Vp = Vt + (size_t)bh * HD * SS;

    int qBase  = qt * 16;
    int kmax32 = (qBase + 16 + 31) & ~31;      // causal extent, padded to 32
    int nkb16  = kmax32 >> 4;
    int ml = lane & 15;
    int sel  = (lane & 16) ? 8 : 0;
    int nsel = (lane & 16) ? 16 : 0;

    // Q A-operand (scale already folded in)
    const __bf16* qrow = Q + (size_t)(qBase + ml) * HD;
    bf16x16 qa0 = cat16(ld8(qrow + nsel),      ld8(qrow + nsel + 8));  // d 0..31
    bf16x16 qa1 = cat16(ld8(qrow + 32 + nsel), ld8(qrow + 40 + nsel));  // d 32..63

    // Phase 1: scores = Q K^T (masked), striped over waves by 16-key tiles
    f32x8 zero = {0.f,0.f,0.f,0.f,0.f,0.f,0.f,0.f};
    for (int kb = wave; kb < nkb16; kb += 8) {
        int key = kb * 16 + ml;
        const __bf16* krow = Kp + (size_t)key * HD;
        f32x8 c = zero;
        c = wmma_bf16(qa0, cat16(ld8(krow + nsel),      ld8(krow + nsel + 8)),  c);
        c = wmma_bf16(qa1, cat16(ld8(krow + 32 + nsel), ld8(krow + 40 + nsel)), c);
        for (int v = 0; v < 8; ++v) {
            int q = qBase + v + sel;
            scores[(v + sel) * SS + key] = (key <= q) ? c[v] : -1e9f;
        }
    }
    __syncthreads();

    // Phase 2: row softmax (2 rows per wave, lanes stride columns)
    for (int rr = 0; rr < 2; ++rr) {
        int r = wave * 2 + rr;
        float* row = scores + (size_t)r * SS;
        float m = -3.4e38f;
        for (int c = lane; c < kmax32; c += 32) m = fmaxf(m, row[c]);
        for (int off = 16; off; off >>= 1) m = fmaxf(m, __shfl_xor(m, off, 32));
        float s = 0.f;
        for (int c = lane; c < kmax32; c += 32) {
            float e = __expf(row[c] - m); row[c] = e; s += e;
        }
        for (int off = 16; off; off >>= 1) s += __shfl_xor(s, off, 32);
        float inv = 1.f / s;
        for (int c = lane; c < kmax32; c += 32) row[c] *= inv;
    }
    __syncthreads();

    // Phase 3: P @ V — 4 dim-tiles x 2-way split-K across 8 waves
    int d0 = (wave & 3) * 16, half = wave >> 2;
    f32x8 acc = zero;
    int nkc = kmax32 >> 5;
    const float*  prow = scores + (size_t)ml * SS;
    const __bf16* vrow = Vp + (size_t)(d0 + ml) * SS + nsel;
    for (int kc = half; kc < nkc; kc += 2) {
        int k0 = kc * 32;
        bf16x16 pa;
        const float* p0 = prow + k0 + nsel;
        const float* p1 = prow + k0 + nsel + 8;
#pragma unroll
        for (int i = 0; i < 8; ++i) {
            pa[i]     = (__bf16)p0[i];
            pa[8 + i] = (__bf16)p1[i];
        }
        acc = wmma_bf16(pa, cat16(ld8(vrow + k0), ld8(vrow + k0 + 8)), acc);
    }
    float* myred = red + half * (16 * HD);
    for (int v = 0; v < 8; ++v) myred[(v + sel) * HD + d0 + ml] = acc[v];
    __syncthreads();

    if (half == 0) {
        int bi = bh >> 4, h = bh & 15;
        for (int v = 0; v < 8; ++v) {
            int r = v + sel;
            float val = red[r * HD + d0 + ml] + red[16 * HD + r * HD + d0 + ml];
            Ab[(size_t)(bi * SS + qBase + r) * NST + h * HD + d0 + ml] = (__bf16)val;
        }
    }
}

// -------------------------------------------------------------- output GEMM
__global__ void __launch_bounds__(256) proj_gemm(
    const __bf16* __restrict__ Ab, const __bf16* __restrict__ WpT,
    const float* __restrict__ bp, float* __restrict__ out) {
    int lane = threadIdx.x & 31, wave = threadIdx.x >> 5;
    int rb = blockIdx.x / (NST / 64);         // 16 row blocks of 256
    int ct = blockIdx.x % (NST / 64);         // 16 col panels
    int rowBase = rb * 256 + wave * 32, colBase = ct * 64;
    f32x8 zero = {0.f,0.f,0.f,0.f,0.f,0.f,0.f,0.f};
    f32x8 acc[8] = {zero, zero, zero, zero, zero, zero, zero, zero};
    gemm_tile_32x64(Ab, WpT, NST, rowBase, colBase, lane, acc);

    int msel = (lane & 16) ? 8 : 0;
    for (int t = 0; t < 4; ++t) {
        int col = colBase + t * 16 + (lane & 15);
        float bias = bp[col];
        for (int g = 0; g < 2; ++g) {
            for (int v = 0; v < 8; ++v) {
                int row = rowBase + g * 16 + v + msel;
                out[(size_t)row * NST + col] = acc[g * 4 + t][v] + bias;
            }
        }
    }
}

// ------------------------------------------------------------------ launcher
extern "C" void kernel_launch(void* const* d_in, const int* in_sizes, int n_in,
                              void* d_out, int out_size, void* d_ws, size_t ws_size,
                              hipStream_t stream) {
    const float* x   = (const float*)d_in[0];
    const float* w_c = (const float*)d_in[1];
    const float* b_c = (const float*)d_in[2];
    const float* w_p = (const float*)d_in[3];
    const float* b_p = (const float*)d_in[4];
    float* out = (float*)d_out;

    // Workspace layout (48 MB total)
    char* ws = (char*)d_ws;
    __bf16* Xb  = (__bf16*)(ws);                          //  8 MB  [4096,1024]
    __bf16* WcT = (__bf16*)(ws + ((size_t) 8 << 20));     //  6 MB  [3072,1024]
    __bf16* WpT = (__bf16*)(ws + ((size_t)14 << 20));     //  2 MB  [1024,1024]
    __bf16* Qb  = (__bf16*)(ws + ((size_t)16 << 20));     //  8 MB  [b,h,s,d]
    __bf16* Kb  = (__bf16*)(ws + ((size_t)24 << 20));     //  8 MB  [b,h,s,d]
    __bf16* Vt  = (__bf16*)(ws + ((size_t)32 << 20));     //  8 MB  [b,h,d,s]
    __bf16* Ab  = (__bf16*)(ws + ((size_t)40 << 20));     //  8 MB  [4096,1024]

    // 1. fp32 -> bf16 conversions / weight transposes
    int n4 = (ROWS * NXX) / 4;
    cvt_f32_to_bf16_x4<<<(n4 + 255) / 256, 256, 0, stream>>>(x, Xb, n4);
    transpose_to_bf16<<<dim3(NC / 32, NXX / 32), dim3(32, 8), 0, stream>>>(w_c, WcT, NXX, NC);
    transpose_to_bf16<<<dim3(NST / 32, NXX / 32), dim3(32, 8), 0, stream>>>(w_p, WpT, NXX, NST);

    // 2. QKV projection: 16 row-blocks x 48 col panels
    qkv_gemm<<<16 * (NC / 64), 256, 0, stream>>>(Xb, WcT, b_c, Qb, Kb, Vt);

    // 3. causal attention (needs >64KB dynamic LDS; CDNA5 WGP has 320KB)
    hipFuncSetAttribute(reinterpret_cast<const void*>(attn_kernel),
                        hipFuncAttributeMaxDynamicSharedMemorySize, ATTN_SMEM);
    attn_kernel<<<BB * NH * (SS / 16), 256, ATTN_SMEM, stream>>>(Qb, Kb, Vt, Ab);

    // 4. output projection: 16 row-blocks x 16 col panels
    proj_gemm<<<16 * (NST / 64), 256, 0, stream>>>(Ab, WpT, b_p, out);
}
